// ReduceState_84043920048454
// MI455X (gfx1250) — matrix operand order repaired and checked by reading, not compiled, not run on hardware
//
#include <hip/hip_runtime.h>
#include <hip/hip_bf16.h>

#define Bb   8
#define Ss   2048
#define Hh   256
#define Ee   128
#define DIN  768          // 2H + 2E
#define G4   1024         // 4H
#define MS   (Bb * Ss)    // 16384
#define LDSN 128          // Whh rows cached in LDS by the TDM

typedef __bf16 bf16_t;
typedef bf16_t v16bf __attribute__((ext_vector_type(16)));
typedef float  v8f   __attribute__((ext_vector_type(8)));
typedef unsigned int v4u __attribute__((ext_vector_type(4)));
typedef int v8i __attribute__((ext_vector_type(8)));
typedef int v4i __attribute__((ext_vector_type(4)));

struct U16x16 { unsigned short u[16]; };
struct U4x2   { uint4 a, b; };

__device__ __forceinline__ unsigned short f2bf(float f) {
  unsigned int x = __builtin_bit_cast(unsigned int, f);
  unsigned int r = x + 0x7FFFu + ((x >> 16) & 1u);   // round-to-nearest-even
  return (unsigned short)(r >> 16);
}

__device__ __forceinline__ float fast_rcp(float x) {
#if __has_builtin(__builtin_amdgcn_rcpf)
  return __builtin_amdgcn_rcpf(x);
#else
  return 1.0f / x;
#endif
}
__device__ __forceinline__ float fast_tanh(float x) {
#if __has_builtin(__builtin_amdgcn_tanh_f32)
  return __builtin_amdgcn_tanh_f32(x);            // CDNA5 V_TANH_F32
#else
  float t = __expf(2.0f * x);
  return 1.0f - 2.0f * fast_rcp(t + 1.0f);
#endif
}
__device__ __forceinline__ float sigm(float x) { return fast_rcp(1.0f + __expf(-x)); }

// A-matrix (16x32, MxK, bf16) K index for VGPR v, lane-half h: contiguous k = 16h + 2v (matches load_b16)
__device__ __forceinline__ int a_kmap(int v, int half) {
  return (half << 4) + (v << 1);
}

// Load A fragment (16x32 tile) from an LDS ushort array with row stride ldk.
__device__ __forceinline__ v16bf load_a_lds(const unsigned short* sm, int ldk, int k0) {
  int lane = threadIdx.x & 31;
  int m = lane & 15, half = lane >> 4;
  U16x16 t;
#pragma unroll
  for (int v = 0; v < 8; ++v) {
    int k = k0 + a_kmap(v, half);
    unsigned int pair = *reinterpret_cast<const unsigned int*>(&sm[m * ldk + k]);
    t.u[2 * v]     = (unsigned short)(pair & 0xFFFFu);
    t.u[2 * v + 1] = (unsigned short)(pair >> 16);
  }
  return __builtin_bit_cast(v16bf, t);
}

// Load B fragment (32x16 = KxN) from a weight array stored [N,K] row-major
// (global or LDS -- addrspace inferred): lane = n, 16 consecutive K per lane.
__device__ __forceinline__ v16bf load_b16(const unsigned short* Bw, int ldk,
                                          int nBase, int k0) {
  int lane = threadIdx.x & 31;
  int n = nBase + (lane & 15);
  int k = k0 + ((lane >> 4) << 4);
  const uint4* p = reinterpret_cast<const uint4*>(Bw + (size_t)n * ldk + k);
  U4x2 raw; raw.a = p[0]; raw.b = p[1];
  return __builtin_bit_cast(v16bf, raw);
}

// Async global->LDS copy, 8 bytes per lane (ASYNCcnt-tracked, CDNA5).
__device__ __forceinline__ void async_copy_b64(unsigned lds_off, const void* gaddr) {
  asm volatile("global_load_async_to_lds_b64 %0, %1, off"
               :: "v"(lds_off), "v"(gaddr) : "memory");
}
__device__ __forceinline__ void wait_asynccnt0() {
  asm volatile("s_wait_asynccnt 0x0" ::: "memory");
}

// ---------------- small prep kernels ----------------

__global__ void k_f2bf(const float* __restrict__ in, unsigned short* __restrict__ out, int n) {
  int i = blockIdx.x * blockDim.x + threadIdx.x;
  if (i < n) out[i] = f2bf(in[i]);
}

__global__ void k_arc_init(const int* __restrict__ heads, const int* __restrict__ rels,
                           const int* __restrict__ posr,
                           const float* __restrict__ rel_emb, const float* __restrict__ pos_emb,
                           float* __restrict__ arcpos) {
  int bs = blockIdx.x, e = threadIdx.x;
  int h = heads[bs];
  float scale = (h > 0) ? 0.5f : 1.0f;
  float re = rel_emb[(size_t)rels[bs] * Ee + e];
  float pe = pos_emb[(size_t)posr[bs] * Ee + e];
  arcpos[(size_t)bs * Ee + e] = re * scale + pe;
}

__global__ void k_arc_scatter(const int* __restrict__ heads, const int* __restrict__ rels,
                              const float* __restrict__ rel_emb, float* __restrict__ arcpos) {
  int bs = blockIdx.x, e = threadIdx.x;
  int h = heads[bs];
  if (h > 0) {
    int b = bs / Ss;
    float re = rel_emb[(size_t)rels[bs] * Ee + e];
    atomicAdd(&arcpos[((size_t)b * Ss + (h - 1)) * Ee + e], 0.5f * re);
  }
}

__global__ void k_pack_x(const float* __restrict__ class_out, const int* __restrict__ text,
                         const float* __restrict__ emb, const float* __restrict__ arcpos,
                         unsigned short* __restrict__ X) {
  long i = (long)blockIdx.x * blockDim.x + threadIdx.x;
  if (i >= (long)MS * DIN) return;
  int col = (int)(i % DIN);
  long row = i / DIN;
  float v;
  if (col < 2 * Hh)            v = class_out[row * (2 * Hh) + col];
  else if (col < 2 * Hh + Ee)  v = emb[(size_t)text[row] * Ee + (col - 2 * Hh)];
  else                         v = arcpos[row * Ee + (col - 2 * Hh - Ee)];
  X[i] = f2bf(v);
}

// ---------------- generic bf16 WMMA GEMM: C[M,N] = A[M,K] * Bw[N,K]^T + bias ----------------
// block = 128 (4 waves), macro-tile 16(M) x 128(N), 2 N-tiles per wave.
// A tile double-buffered in LDS, filled via async-to-LDS; one barrier per K-step.

__global__ __launch_bounds__(128) void k_gemm_bf16(
    const unsigned short* __restrict__ A, const unsigned short* __restrict__ Bw,
    const float* __restrict__ bias1, const float* __restrict__ bias2,
    float* __restrict__ C, int M, int N, int K) {
  __shared__ unsigned short sA[2][16 * 32];
  int wave = threadIdx.x >> 5, lane = threadIdx.x & 31;
  int mBase = blockIdx.y * 16;
  int nBase0 = blockIdx.x * 128 + wave * 32;

  auto stage = [&](int buf, int k0) {
    int m = threadIdx.x >> 3, c = (threadIdx.x & 7) << 2;  // 8 bytes / thread
    unsigned lds_off = (unsigned)(size_t)&sA[buf][m * 32 + c];
    async_copy_b64(lds_off, &A[(size_t)(mBase + m) * K + k0 + c]);
  };

  v8f acc0 = {}, acc1 = {};
  stage(0, 0);
  int nIter = K / 32;
  for (int i = 0; i < nIter; ++i) {
    int k0 = i * 32;
    wait_asynccnt0();
    __syncthreads();
    if (i + 1 < nIter) stage((i + 1) & 1, k0 + 32);
    if (i + 2 < nIter) {
      __builtin_prefetch(&Bw[(size_t)(nBase0 + (lane & 15)) * K + k0 + 64], 0, 1);
      __builtin_prefetch(&Bw[(size_t)(nBase0 + 16 + (lane & 15)) * K + k0 + 64], 0, 1);
    }
    v16bf a  = load_a_lds(sA[i & 1], 32, 0);
    v16bf b0 = load_b16(Bw, K, nBase0, k0);
    v16bf b1 = load_b16(Bw, K, nBase0 + 16, k0);
    acc0 = __builtin_amdgcn_wmma_f32_16x16x32_bf16(false, a, false, b0, (short)0, acc0,
                                                   false, false);
    acc1 = __builtin_amdgcn_wmma_f32_16x16x32_bf16(false, a, false, b1, (short)0, acc1,
                                                   false, false);
  }
  int half = lane >> 4, nc = lane & 15;
#pragma unroll
  for (int tile = 0; tile < 2; ++tile) {
    int n = nBase0 + tile * 16 + nc;
    float bsum = (bias1 ? bias1[n] : 0.0f) + (bias2 ? bias2[n] : 0.0f);
    const v8f& acc = tile ? acc1 : acc0;
#pragma unroll
    for (int r = 0; r < 8; ++r) {
      int m = mBase + r + half * 8;
      C[(size_t)m * N + n] = acc[r] + bsum;
    }
  }
}

// ---------------- persistent bidirectional LSTM recurrence ----------------
// grid.x = 2 (directions), block = 1024 (32 waves). Per step:
//   g = h @ Whh^T via WMMA (M=16 padded, N=1024, K=256), + pipelined x-gates,
//   then elementwise cell update; h in LDS (bf16), c in registers.
// Whh rows [0,128) parked in LDS once via the Tensor Data Mover.

__global__ __launch_bounds__(1024) void k_lstm(
    const float* __restrict__ gates_f, const float* __restrict__ gates_b,
    const unsigned short* __restrict__ Whh_f, const unsigned short* __restrict__ Whh_b,
    const float* __restrict__ h0, const float* __restrict__ c0,
    float* __restrict__ out_f, float* __restrict__ out_b,
    float* __restrict__ hT, float* __restrict__ cT) {
  int dir = blockIdx.x;
  const float*          gates = dir ? gates_b : gates_f;
  const unsigned short* Whh   = dir ? Whh_b   : Whh_f;
  float*                out   = dir ? out_b   : out_f;

  __shared__ unsigned short hS[16 * Hh];        // 8 KB, rows 8..15 stay zero (M pad)
  __shared__ float          gS[Bb * G4];        // 32 KB
  __shared__ unsigned short sWhh[LDSN * Hh];    // 64 KB, Whh rows [0,128)

  int tid = threadIdx.x;
  for (int i = tid; i < 16 * Hh; i += 1024) {
    int m = i >> 8, k = i & 255;
    float v = (m < Bb) ? h0[dir * Bb * Hh + m * Hh + k] : 0.0f;
    hS[i] = f2bf(v);
  }

  // TDM: one wave DMAs Whh[0:128, 0:256] (bf16) into sWhh. 2D descriptor, ISA S8.3-8.5.
  if (tid == 0) {
    unsigned lds_off = (unsigned)(size_t)&sWhh[0];
    unsigned long long ga = (unsigned long long)(const void*)Whh;
    v4u g0 = { 1u,                                   // count=1 (user D#)
               lds_off,                              // lds_addr
               (unsigned)(ga & 0xFFFFFFFFull),
               (unsigned)((ga >> 32) & 0x1FFFFFFull) | 0x80000000u };  // addr[56:32]|type=2
    v8i g1 = { 0x00010000,                           // data_size=2B, wg_mask=0
               (int)(256u << 16),                    // tensor_dim0 = 256 (bits 79:48 lo)
               (int)(1024u << 16),                   // dim0 hi=0 | tensor_dim1 = 1024
               (int)(256u << 16),                    // dim1 hi=0 | tile_dim0 = 256
               LDSN,                                 // tile_dim1 = 128 | tile_dim2 = 0
               256, 0, 0 };                          // tensor_dim0_stride = 256
    v4i gz = { 0, 0, 0, 0 };
#if defined(__clang_major__) && __clang_major__ >= 23
    v8i gz8 = { 0, 0, 0, 0, 0, 0, 0, 0 };
    __builtin_amdgcn_tensor_load_to_lds(g0, g1, gz, gz, gz8, 0);
#else
    __builtin_amdgcn_tensor_load_to_lds(g0, g1, gz, gz, 0);
#endif
    __builtin_amdgcn_s_wait_tensorcnt(0);
  }

  int e0 = tid * 2;                                  // this thread owns cells e0, e0+1
  int brow = e0 >> 8, j0 = e0 & 255;                 // j0 even
  float cA = c0[dir * Bb * Hh + e0];
  float cB = c0[dir * Bb * Hh + e0 + 1];
  __syncthreads();

  int wave = tid >> 5, lane = tid & 31;
  int half = lane >> 4, nc = lane & 15;

  // software-pipelined x-gate loads (one step ahead)
  auto gload = [&](int t_x, float2& i2, float2& f2v, float2& g2, float2& o2) {
    size_t gx = ((size_t)(brow * Ss + t_x)) * G4 + j0;
    i2  = *reinterpret_cast<const float2*>(&gates[gx]);
    f2v = *reinterpret_cast<const float2*>(&gates[gx + 256]);
    g2  = *reinterpret_cast<const float2*>(&gates[gx + 512]);
    o2  = *reinterpret_cast<const float2*>(&gates[gx + 768]);
  };
  float2 xi = {0, 0}, xf = {0, 0}, xg = {0, 0}, xo = {0, 0};
  gload(dir ? (Ss - 1) : 0, xi, xf, xg, xo);

  for (int t = 0; t < Ss; ++t) {
    v16bf aF[8];
#pragma unroll
    for (int kc = 0; kc < 8; ++kc) aF[kc] = load_a_lds(hS, Hh, kc * 32);
#pragma unroll
    for (int tile = 0; tile < 2; ++tile) {
      int nBase = (wave * 2 + tile) * 16;
      v8f acc = {};
      if (nBase < LDSN) {                            // wave-uniform branch
#pragma unroll
        for (int kc = 0; kc < 8; ++kc) {
          v16bf b = load_b16(sWhh, Hh, nBase, kc * 32);          // ds_load
          acc = __builtin_amdgcn_wmma_f32_16x16x32_bf16(false, aF[kc], false, b,
                                                        (short)0, acc, false, false);
        }
      } else {
#pragma unroll
        for (int kc = 0; kc < 8; ++kc) {
          v16bf b = load_b16(Whh, Hh, nBase, kc * 32);           // L2-resident stream
          acc = __builtin_amdgcn_wmma_f32_16x16x32_bf16(false, aF[kc], false, b,
                                                        (short)0, acc, false, false);
        }
      }
      if (half == 0) {                               // rows 0..7 are real batches
#pragma unroll
        for (int r = 0; r < 8; ++r) gS[r * G4 + nBase + nc] = acc[r];
      }
    }
    __syncthreads();

    int t_x = dir ? (Ss - 1 - t) : t;
    float2 ni = {0, 0}, nf = {0, 0}, ng = {0, 0}, no = {0, 0};
    if (t + 1 < Ss) gload(dir ? (Ss - 2 - t) : (t + 1), ni, nf, ng, no);

    float2 ri = *reinterpret_cast<const float2*>(&gS[brow * G4 + j0]);
    float2 rf = *reinterpret_cast<const float2*>(&gS[brow * G4 + 256 + j0]);
    float2 rg = *reinterpret_cast<const float2*>(&gS[brow * G4 + 512 + j0]);
    float2 ro = *reinterpret_cast<const float2*>(&gS[brow * G4 + 768 + j0]);
    float gi0 = ri.x + xi.x, gi1 = ri.y + xi.y;
    float gf0 = rf.x + xf.x, gf1 = rf.y + xf.y;
    float gg0 = rg.x + xg.x, gg1 = rg.y + xg.y;
    float go0 = ro.x + xo.x, go1 = ro.y + xo.y;
    cA = sigm(gf0) * cA + sigm(gi0) * fast_tanh(gg0);
    cB = sigm(gf1) * cB + sigm(gi1) * fast_tanh(gg1);
    float h0v = sigm(go0) * fast_tanh(cA);
    float h1v = sigm(go1) * fast_tanh(cB);
    *reinterpret_cast<float2*>(&out[((size_t)(brow * Ss + t_x)) * Hh + j0]) =
        make_float2(h0v, h1v);
    *reinterpret_cast<unsigned*>(&hS[brow * Hh + j0]) =
        (unsigned)f2bf(h0v) | ((unsigned)f2bf(h1v) << 16);
    if (t == Ss - 1) {
      *reinterpret_cast<float2*>(&hT[dir * Bb * Hh + e0]) = make_float2(h0v, h1v);
      *reinterpret_cast<float2*>(&cT[dir * Bb * Hh + e0]) = make_float2(cA, cB);
    }
    xi = ni; xf = nf; xg = ng; xo = no;
    __syncthreads();
  }
}

// ---------------- attention: one wave per (b,s) ----------------

__global__ __launch_bounds__(256) void k_attn(
    const float* __restrict__ out_f, const float* __restrict__ out_b,
    const float* __restrict__ class_out, const float* __restrict__ dp_out,
    const float* __restrict__ pos_out,
    float* __restrict__ cross, unsigned short* __restrict__ cross_bf) {
  int wv = threadIdx.x >> 5, lane = threadIdx.x & 31;
  int pos = blockIdx.x * 8 + wv;
  int d0 = lane * 16;
  float enc[16], c1[16], c2[16], c3[16];
#pragma unroll
  for (int i = 0; i < 16; ++i) {
    int d = d0 + i;
    enc[i] = (d < Hh) ? out_f[(size_t)pos * Hh + d] : out_b[(size_t)pos * Hh + (d - Hh)];
    c1[i] = class_out[(size_t)pos * 512 + d];
    c2[i] = dp_out[(size_t)pos * 512 + d];
    c3[i] = pos_out[(size_t)pos * 512 + d];
  }
  float s0 = 0, s1 = 0, s2 = 0, s3 = 0;
#pragma unroll
  for (int i = 0; i < 16; ++i) {
    s0 += enc[i] * enc[i]; s1 += enc[i] * c1[i];
    s2 += enc[i] * c2[i];  s3 += enc[i] * c3[i];
  }
#pragma unroll
  for (int off = 16; off; off >>= 1) {      // wave32 reduction
    s0 += __shfl_xor(s0, off, 32); s1 += __shfl_xor(s1, off, 32);
    s2 += __shfl_xor(s2, off, 32); s3 += __shfl_xor(s3, off, 32);
  }
  float mx = fmaxf(fmaxf(s0, s1), fmaxf(s2, s3));
  float e0v = __expf(s0 - mx), e1v = __expf(s1 - mx);
  float e2v = __expf(s2 - mx), e3v = __expf(s3 - mx);
  float inv = fast_rcp(e0v + e1v + e2v + e3v);
  float a0 = e0v * inv, a1 = e1v * inv, a2 = e2v * inv, a3 = e3v * inv;
#pragma unroll
  for (int i = 0; i < 16; ++i) {
    float v = a0 * enc[i] + a1 * c1[i] + a2 * c2[i] + a3 * c3[i];
    cross[(size_t)pos * 512 + d0 + i] = v;
    cross_bf[(size_t)pos * 512 + d0 + i] = f2bf(v);
  }
}

// ---------------- init_h / init_c: tiny [8x512]x[512x256] ----------------

__global__ void k_init_hc(const float* __restrict__ hT, const float* __restrict__ cT,
                          const float* __restrict__ rhW, const float* __restrict__ rhb,
                          const float* __restrict__ rcW, const float* __restrict__ rcb,
                          float* __restrict__ outh, float* __restrict__ outc) {
  int i = blockIdx.x * blockDim.x + threadIdx.x;
  if (i >= Bb * Hh) return;
  int b = i >> 8, j = i & 255;
  float ah = rhb[j], ac = rcb[j];
  for (int k = 0; k < 2 * Hh; ++k) {
    float hv = (k < Hh) ? hT[b * Hh + k] : hT[Bb * Hh + b * Hh + (k - Hh)];
    float cv = (k < Hh) ? cT[b * Hh + k] : cT[Bb * Hh + b * Hh + (k - Hh)];
    ah += hv * rhW[j * 512 + k];
    ac += cv * rcW[j * 512 + k];
  }
  outh[i] = fmaxf(ah, 0.0f);
  outc[i] = fmaxf(ac, 0.0f);
}

// ---------------- launcher ----------------

extern "C" void kernel_launch(void* const* d_in, const int* in_sizes, int n_in,
                              void* d_out, int out_size, void* d_ws, size_t ws_size,
                              hipStream_t stream) {
  const int*   inputs_text = (const int*)d_in[0];
  const float* class_out   = (const float*)d_in[1];
  const float* class_h     = (const float*)d_in[2];
  const float* class_c     = (const float*)d_in[3];
  const float* dp_out      = (const float*)d_in[4];
  const int*   dp_heads    = (const int*)d_in[5];
  const int*   dp_rels     = (const int*)d_in[6];
  const float* pos_out     = (const float*)d_in[7];
  const int*   pos_res     = (const int*)d_in[8];
  const float* emb         = (const float*)d_in[9];
  const float* pos_emb     = (const float*)d_in[10];
  const float* rel_emb     = (const float*)d_in[11];
  const float* Wih_f       = (const float*)d_in[12];
  const float* Whh_f       = (const float*)d_in[13];
  const float* bih_f       = (const float*)d_in[14];
  const float* bhh_f       = (const float*)d_in[15];
  const float* Wih_b       = (const float*)d_in[16];
  const float* Whh_b       = (const float*)d_in[17];
  const float* bih_b       = (const float*)d_in[18];
  const float* bhh_b       = (const float*)d_in[19];
  const float* rh_W        = (const float*)d_in[20];
  const float* rh_b        = (const float*)d_in[21];
  const float* rc_W        = (const float*)d_in[22];
  const float* rc_b        = (const float*)d_in[23];
  const float* Wh_W        = (const float*)d_in[24];
  const float* Wh_b        = (const float*)d_in[25];

  char* ws = (char*)d_ws;
  size_t off = 0;
  auto alloc = [&](size_t bytes) { char* p = ws + off; off += (bytes + 255) & ~(size_t)255; return p; };
  unsigned short* X        = (unsigned short*)alloc((size_t)MS * DIN * 2);
  unsigned short* Wihf_bf  = (unsigned short*)alloc((size_t)G4 * DIN * 2);
  unsigned short* Wihb_bf  = (unsigned short*)alloc((size_t)G4 * DIN * 2);
  unsigned short* Whhf_bf  = (unsigned short*)alloc((size_t)G4 * Hh * 2);
  unsigned short* Whhb_bf  = (unsigned short*)alloc((size_t)G4 * Hh * 2);
  unsigned short* WhW_bf   = (unsigned short*)alloc((size_t)512 * 512 * 2);
  float*          arcpos   = (float*)alloc((size_t)MS * Ee * 4);
  float*          gates_f  = (float*)alloc((size_t)MS * G4 * 4);
  float*          gates_b  = (float*)alloc((size_t)MS * G4 * 4);
  float*          out_f    = (float*)alloc((size_t)MS * Hh * 4);
  float*          out_b    = (float*)alloc((size_t)MS * Hh * 4);
  float*          hT       = (float*)alloc((size_t)2 * Bb * Hh * 4);
  float*          cT       = (float*)alloc((size_t)2 * Bb * Hh * 4);
  unsigned short* cross_bf = (unsigned short*)alloc((size_t)MS * 512 * 2);

  float* o_cross = (float*)d_out;                     // [B,S,512]
  float* o_feats = o_cross + (size_t)MS * 512;        // [B*S,512]
  float* o_inith = o_feats + (size_t)MS * 512;        // [1,B,256]
  float* o_initc = o_inith + (size_t)Bb * Hh;         // [1,B,256]

  auto cvt = [&](const float* src, unsigned short* dst, int n) {
    k_f2bf<<<(n + 255) / 256, 256, 0, stream>>>(src, dst, n);
  };
  cvt(Wih_f, Wihf_bf, G4 * DIN);
  cvt(Wih_b, Wihb_bf, G4 * DIN);
  cvt(Whh_f, Whhf_bf, G4 * Hh);
  cvt(Whh_b, Whhb_bf, G4 * Hh);
  cvt(Wh_W,  WhW_bf,  512 * 512);

  k_arc_init<<<MS, Ee, 0, stream>>>(dp_heads, dp_rels, pos_res, rel_emb, pos_emb, arcpos);
  k_arc_scatter<<<MS, Ee, 0, stream>>>(dp_heads, dp_rels, rel_emb, arcpos);
  {
    long total = (long)MS * DIN;
    k_pack_x<<<(unsigned)((total + 255) / 256), 256, 0, stream>>>(class_out, inputs_text,
                                                                  emb, arcpos, X);
  }

  k_gemm_bf16<<<dim3(G4 / 128, MS / 16), 128, 0, stream>>>(X, Wihf_bf, bih_f, bhh_f,
                                                           gates_f, MS, G4, DIN);
  k_gemm_bf16<<<dim3(G4 / 128, MS / 16), 128, 0, stream>>>(X, Wihb_bf, bih_b, bhh_b,
                                                           gates_b, MS, G4, DIN);

  k_lstm<<<2, 1024, 0, stream>>>(gates_f, gates_b, Whhf_bf, Whhb_bf, class_h, class_c,
                                 out_f, out_b, hT, cT);

  k_attn<<<MS / 8, 256, 0, stream>>>(out_f, out_b, class_out, dp_out, pos_out,
                                     o_cross, cross_bf);
  k_gemm_bf16<<<dim3(512 / 128, MS / 16), 128, 0, stream>>>(cross_bf, WhW_bf, Wh_b, nullptr,
                                                            o_feats, MS, 512, 512);
  k_init_hc<<<(Bb * Hh + 255) / 256, 256, 0, stream>>>(hT, cT, rh_W, rh_b, rc_W, rc_b,
                                                       o_inith, o_initc);
}
